// IntelligentNetwork_55585466745306
// MI455X (gfx1250) — matrix operand order repaired and checked by reading, not compiled, not run on hardware
//
#include <hip/hip_runtime.h>
#include <math.h>

// ---- network config (matches reference) ----
#define NI_   128
#define NT_   3584
#define NA_   384
#define NN_   4096          // total neurons
#define HID_  32
#define KQ_   16
#define SIG_  16
#define IND_  128
#define OUTD_ 64
#define BB_   64
#define TT_   32
#define CD_   1024          // BB_*SIG_ : batched "N" dim of the big GEMM

typedef __bf16 bf16;
typedef __attribute__((ext_vector_type(16))) __bf16 v16bf;
typedef __attribute__((ext_vector_type(8)))  __bf16 v8bf;
typedef __attribute__((ext_vector_type(8)))  float  v8f;
typedef __attribute__((ext_vector_type(4)))  unsigned int u32x4;

__device__ __forceinline__ unsigned short f2bf(float f) {
    unsigned int u = __float_as_uint(f);
    u += 0x7FFFu + ((u >> 16) & 1u);       // round-to-nearest-even
    return (unsigned short)(u >> 16);
}
__device__ __forceinline__ float bf2f(unsigned short h) {
    return __uint_as_float(((unsigned int)h) << 16);
}

// ------------------------------------------------------------------
// attn = softmax(q @ k^T / sqrt(KQ)) ; one block per row, bf16 output
// ------------------------------------------------------------------
__global__ __launch_bounds__(256)
void attn_softmax_kernel(const float* __restrict__ q_all,
                         const float* __restrict__ k_all,
                         unsigned short* __restrict__ attn) {
    const int n   = blockIdx.x;
    const int tid = threadIdx.x;
    __shared__ float qrow[KQ_];
    __shared__ float red[256];
    if (tid < KQ_) qrow[tid] = q_all[n * KQ_ + tid];
    __syncthreads();

    float s[16];
    float lmax = -3.4e38f;
#pragma unroll
    for (int j = 0; j < 16; ++j) {
        const int m = tid + j * 256;
        const float* kr = k_all + (size_t)m * KQ_;
        float d = 0.f;
#pragma unroll
        for (int t = 0; t < KQ_; ++t) d += qrow[t] * kr[t];
        d *= 0.25f;                 // 1/sqrt(16)
        s[j] = d;
        lmax = fmaxf(lmax, d);
    }
    red[tid] = lmax; __syncthreads();
    for (int off = 128; off > 0; off >>= 1) {
        if (tid < off) red[tid] = fmaxf(red[tid], red[tid + off]);
        __syncthreads();
    }
    const float mx = red[0];
    __syncthreads();

    float lsum = 0.f;
#pragma unroll
    for (int j = 0; j < 16; ++j) { s[j] = __expf(s[j] - mx); lsum += s[j]; }
    red[tid] = lsum; __syncthreads();
    for (int off = 128; off > 0; off >>= 1) {
        if (tid < off) red[tid] += red[tid + off];
        __syncthreads();
    }
    const float inv = 1.0f / red[0];
#pragma unroll
    for (int j = 0; j < 16; ++j)
        attn[(size_t)n * NN_ + tid + j * 256] = f2bf(s[j] * inv);
}

// ------------------------------------------------------------------
// zero-fill (initial recurrent state)
// ------------------------------------------------------------------
__global__ __launch_bounds__(256)
void zero_kernel(unsigned int* __restrict__ p, int nwords) {
    const int i = blockIdx.x * 256 + threadIdx.x;
    if (i < nwords) p[i] = 0u;
}

// ------------------------------------------------------------------
// ctx[n, c] = sum_m attn[n,m] * stateT[c, m]       (c = b*SIG + d)
// LDS-blocked WMMA bf16 GEMM: M=3968, N=1024, K=4096.
//   workgroup = 512 threads (16 waves) -> 128x128 macro tile
//   K-chunk 32, double-buffered LDS (padded rows: 40 elems to spread banks)
//   wave -> 32x32 sub-tile: 2 A-frags x 2 B-frags -> 4 wmma / K-step
// A frag (16x32 bf16): lane row = l&15, K chunks at (l>>4)*8 and 16+(l>>4)*8
// B frag (32x16 bf16): lane col = l&15, K chunks at (l>>4)*8 and 16+(l>>4)*8
// ------------------------------------------------------------------
#define LDS_PITCH 40   // 32 k-elements + 8 pad (80B rows, 16B aligned)

__global__ __launch_bounds__(512)
void ctx_gemm_kernel(const unsigned short* __restrict__ attn,
                     const unsigned short* __restrict__ stT,
                     float* __restrict__ ctx) {
    __shared__ __align__(16) unsigned short ldsA[2][128 * LDS_PITCH];
    __shared__ __align__(16) unsigned short ldsB[2][128 * LDS_PITCH];

    const int tid  = threadIdx.x;
    const int lane = tid & 31;
    const int w    = tid >> 5;            // 0..15
    const int mrow = lane & 15;
    const int hh   = lane >> 4;
    const int n0   = NI_ + blockIdx.y * 128;
    const int c0   = blockIdx.x * 128;
    const int rw0  = (w >> 2) * 32;       // wave row offset in macro tile
    const int cw0  = (w & 3) * 32;        // wave col offset in macro tile

    // staging: each thread moves one 16B chunk of A and one of B per K-chunk
    const int ar = tid >> 2;              // 0..127 : row (A) / col (B) in tile
    const int ak = (tid & 3) * 8;         // element offset inside 32-k chunk
    const unsigned short* gA = attn + (size_t)(n0 + ar) * NN_ + ak;
    const unsigned short* gB = stT  + (size_t)(c0 + ar) * NN_ + ak;
    const int lofs = ar * LDS_PITCH + ak; // LDS element offset

    // prologue: chunk 0 -> LDS buf0, chunk 1 -> regs
    u32x4 regA = *(const u32x4*)(gA);
    u32x4 regB = *(const u32x4*)(gB);
    *(u32x4*)&ldsA[0][lofs] = regA;
    *(u32x4*)&ldsB[0][lofs] = regB;
    regA = *(const u32x4*)(gA + 32);
    regB = *(const u32x4*)(gB + 32);
    __syncthreads();

    v8f acc[2][2];
#pragma unroll
    for (int rt = 0; rt < 2; ++rt)
#pragma unroll
        for (int ct = 0; ct < 2; ++ct)
#pragma unroll
            for (int v = 0; v < 8; ++v) acc[rt][ct][v] = 0.f;

    const int NIT = NN_ / 32;             // 128 K-steps
    for (int it = 0; it < NIT; ++it) {
        const int cur = it & 1;

        // fragments out of LDS
        v16bf afr[2], bfr[2];
#pragma unroll
        for (int rt = 0; rt < 2; ++rt) {
            const unsigned short* base =
                &ldsA[cur][(rw0 + rt * 16 + mrow) * LDS_PITCH];
            const v8bf lo = *(const v8bf*)(base + hh * 8);
            const v8bf hi = *(const v8bf*)(base + 16 + hh * 8);
#pragma unroll
            for (int i = 0; i < 8; ++i) { afr[rt][i] = lo[i]; afr[rt][i + 8] = hi[i]; }
        }
#pragma unroll
        for (int ct = 0; ct < 2; ++ct) {
            const unsigned short* base =
                &ldsB[cur][(cw0 + ct * 16 + mrow) * LDS_PITCH];
            const v8bf lo = *(const v8bf*)(base + hh * 8);
            const v8bf hi = *(const v8bf*)(base + 16 + hh * 8);
#pragma unroll
            for (int i = 0; i < 8; ++i) { bfr[ct][i] = lo[i]; bfr[ct][i + 8] = hi[i]; }
        }

        // stage next chunk into the other buffer; prefetch the one after
        if (it + 1 < NIT) {
            *(u32x4*)&ldsA[cur ^ 1][lofs] = regA;
            *(u32x4*)&ldsB[cur ^ 1][lofs] = regB;
        }
        if (it + 2 < NIT) {
            regA = *(const u32x4*)(gA + (size_t)(it + 2) * 32);
            regB = *(const u32x4*)(gB + (size_t)(it + 2) * 32);
        }

#pragma unroll
        for (int rt = 0; rt < 2; ++rt)
#pragma unroll
            for (int ct = 0; ct < 2; ++ct)
                acc[rt][ct] = __builtin_amdgcn_wmma_f32_16x16x32_bf16(
                                  false, afr[rt], false, bfr[ct],
                                  (short)0, acc[rt][ct], false, false);

        __syncthreads();   // waits DScnt: buffers safe to flip
    }

    // C/D layout: VGPR v, lane l -> row v + (l>>4)*8, col l&15
#pragma unroll
    for (int rt = 0; rt < 2; ++rt)
#pragma unroll
        for (int ct = 0; ct < 2; ++ct)
#pragma unroll
            for (int v = 0; v < 8; ++v)
                ctx[(size_t)(n0 + rw0 + rt * 16 + hh * 8 + v) * CD_
                    + c0 + cw0 + ct * 16 + mrow] = acc[rt][ct][v];
}

// ------------------------------------------------------------------
// transmission + action neurons: out = W2 @ tanh(W1 @ ctx + b1) + b2
// block = neuron (weights lane-uniform -> broadcast), thread = batch
// writes transposed bf16 state: stateT[(b*SIG+d)*NN + n]
// ------------------------------------------------------------------
__global__ __launch_bounds__(64)
void mlp_o_kernel(const float* __restrict__ ctx,
                  const float* __restrict__ W1, const float* __restrict__ b1,
                  const float* __restrict__ W2, const float* __restrict__ b2,
                  unsigned short* __restrict__ stNew) {
    const int idx = blockIdx.x;        // 0 .. NT_+NA_-1
    const int n   = NI_ + idx;
    const int b   = threadIdx.x;       // 0 .. 63

    float c[SIG_];
    const float* cr = ctx + (size_t)n * CD_ + b * SIG_;
#pragma unroll
    for (int s = 0; s < SIG_; ++s) c[s] = cr[s];

    float h[HID_];
    const float* w1 = W1 + (size_t)idx * HID_ * SIG_;
#pragma unroll 4
    for (int j = 0; j < HID_; ++j) {
        float a = b1[idx * HID_ + j];
#pragma unroll
        for (int s = 0; s < SIG_; ++s) a += w1[j * SIG_ + s] * c[s];
        h[j] = tanhf(a);
    }
    const float* w2 = W2 + (size_t)idx * SIG_ * HID_;
#pragma unroll
    for (int d = 0; d < SIG_; ++d) {
        float a = b2[idx * SIG_ + d];
#pragma unroll 8
        for (int j = 0; j < HID_; ++j) a += w2[d * HID_ + j] * h[j];
        stNew[(size_t)(b * SIG_ + d) * NN_ + n] = f2bf(a);
    }
}

// ------------------------------------------------------------------
// input neurons: consume raw x_t instead of attention context
// ------------------------------------------------------------------
__global__ __launch_bounds__(64)
void mlp_in_kernel(const float* __restrict__ x, int t,
                   const float* __restrict__ W1, const float* __restrict__ b1,
                   const float* __restrict__ W2, const float* __restrict__ b2,
                   unsigned short* __restrict__ stNew) {
    const int n = blockIdx.x;          // 0 .. NI_-1
    const int b = threadIdx.x;         // 0 .. 63
    const float* xr = x + ((size_t)b * TT_ + t) * IND_;
    const float* w1 = W1 + (size_t)n * HID_ * IND_;

    float h[HID_];
    for (int j = 0; j < HID_; ++j) {
        float a = b1[n * HID_ + j];
#pragma unroll 16
        for (int i = 0; i < IND_; ++i) a += w1[j * IND_ + i] * xr[i];
        h[j] = tanhf(a);
    }
    const float* w2 = W2 + (size_t)n * SIG_ * HID_;
#pragma unroll
    for (int d = 0; d < SIG_; ++d) {
        float a = b2[n * SIG_ + d];
#pragma unroll 8
        for (int j = 0; j < HID_; ++j) a += w2[d * HID_ + j] * h[j];
        stNew[(size_t)(b * SIG_ + d) * NN_ + n] = f2bf(a);
    }
}

// ------------------------------------------------------------------
// y[b, t, o] = bp[o] + act[b,:] . Wp[o,:]   (action slice of new state)
// block = batch (state loads lane-uniform), thread = output channel
// ------------------------------------------------------------------
__global__ __launch_bounds__(64)
void out_proj_kernel(const unsigned short* __restrict__ st,
                     const float* __restrict__ Wp, const float* __restrict__ bp,
                     float* __restrict__ out, int t) {
    const int b = blockIdx.x;
    const int o = threadIdx.x;
    float acc = bp[o];
    const float* wr = Wp + (size_t)o * (NA_ * SIG_);
    for (int np = 0; np < NA_; ++np) {
        const int col = NI_ + NT_ + np;
#pragma unroll
        for (int d = 0; d < SIG_; ++d) {
            const float a = bf2f(st[(size_t)(b * SIG_ + d) * NN_ + col]);
            acc += a * wr[np * SIG_ + d];
        }
    }
    out[((size_t)b * TT_ + t) * OUTD_ + o] = acc;
}

// ------------------------------------------------------------------
extern "C" void kernel_launch(void* const* d_in, const int* in_sizes, int n_in,
                              void* d_out, int out_size, void* d_ws, size_t ws_size,
                              hipStream_t stream) {
    const float* x     = (const float*)d_in[0];
    const float* W1_in = (const float*)d_in[1];
    const float* b1_in = (const float*)d_in[2];
    const float* W2_in = (const float*)d_in[3];
    const float* b2_in = (const float*)d_in[4];
    const float* W1_o  = (const float*)d_in[5];
    const float* b1_o  = (const float*)d_in[6];
    const float* W2_o  = (const float*)d_in[7];
    const float* b2_o  = (const float*)d_in[8];
    const float* q_all = (const float*)d_in[9];
    const float* k_all = (const float*)d_in[10];
    const float* Wp    = (const float*)d_in[11];
    const float* bp    = (const float*)d_in[12];
    float* out = (float*)d_out;

    // workspace carve-up (all resident in 192 MB L2):
    //   attn bf16  : 32 MB | state x2 bf16 (transposed [CD_, NN_]) : 16 MB | ctx f32 : 16 MB
    char* ws = (char*)d_ws;
    size_t off = 0;
    unsigned short* attn = (unsigned short*)(ws + off); off += (size_t)NN_ * NN_ * 2;
    unsigned short* st0  = (unsigned short*)(ws + off); off += (size_t)CD_ * NN_ * 2;
    unsigned short* st1  = (unsigned short*)(ws + off); off += (size_t)CD_ * NN_ * 2;
    float*          ctx  = (float*)(ws + off);          off += (size_t)NN_ * CD_ * 4;

    // attention weights are time/batch independent: compute once
    attn_softmax_kernel<<<NN_, 256, 0, stream>>>(q_all, k_all, attn);

    // prev0 = zeros
    const int nwords = (CD_ * NN_) / 2;   // bf16 elems / 2 per u32
    zero_kernel<<<(nwords + 255) / 256, 256, 0, stream>>>((unsigned int*)st0, nwords);

    for (int t = 0; t < TT_; ++t) {
        unsigned short* prev = (t & 1) ? st1 : st0;
        unsigned short* next = (t & 1) ? st0 : st1;
        // 128x128 macro tiles: cols 1024/128 = 8, rows 3968/128 = 31
        ctx_gemm_kernel<<<dim3(8, 31), 512, 0, stream>>>(attn, prev, ctx);
        mlp_in_kernel<<<NI_, 64, 0, stream>>>(x, t, W1_in, b1_in, W2_in, b2_in, next);
        mlp_o_kernel<<<NT_ + NA_, 64, 0, stream>>>(ctx, W1_o, b1_o, W2_o, b2_o, next);
        out_proj_kernel<<<BB_, 64, 0, stream>>>(next, Wp, bp, out, t);
    }
    (void)in_sizes; (void)n_in; (void)out_size; (void)ws_size;
}
